// MultiHeadAttention_32126355374662
// MI455X (gfx1250) — matrix operand order repaired and checked by reading, not compiled, not run on hardware
//
#include <hip/hip_runtime.h>
#include <hip/hip_bf16.h>

#define D_EMBED 768
#define N_HEADS 12
#define HD      64
#define NB      2
#define SQ      2048

typedef _Float16 v16h __attribute__((ext_vector_type(16)));
typedef _Float16 v8h  __attribute__((ext_vector_type(8)));
typedef _Float16 v4h  __attribute__((ext_vector_type(4)));
typedef float    v8f  __attribute__((ext_vector_type(8)));
typedef float    v4f  __attribute__((ext_vector_type(4)));

static __device__ __forceinline__ v8f wmma_f16(v16h a, v16h b, v8f c) {
  // D = A(16x32 f16) * B(32x16 f16) + C(16x16 f32)
  return __builtin_amdgcn_wmma_f32_16x16x32_f16(false, a, false, b, (short)0, c,
                                                false, false);
}

static __device__ __forceinline__ v8f vzero8() {
  v8f z = {0.f, 0.f, 0.f, 0.f, 0.f, 0.f, 0.f, 0.f};
  return z;
}

// A fragment (16x32 f16), row-major src [*, ld].
// lane: row m0+lane%16; K elems e0..7 = kA..kA+7, e8..15 = kA+16..kA+23,
// kA = k0 + 8*(lane/16).   (ISA 16-bit A-matrix table)
static __device__ __forceinline__ v16h a_frag_f16(const _Float16* base, int ld,
                                                  int m0, int k0, int lane) {
  int m  = m0 + (lane & 15);
  int kA = k0 + 8 * (lane >> 4);
  const _Float16* p = base + (size_t)m * ld + kA;
  v8h lo = *(const v8h*)(p);
  v8h hi = *(const v8h*)(p + 16);
  v16h r;
#pragma unroll
  for (int i = 0; i < 8; ++i) {
    r[i]     = lo[i];
    r[8 + i] = hi[i];
  }
  return r;
}

// B fragment (32x16 f16) from row-major W[n][k] (C = A * W^T).
// lane: column n0+lane%16; K elems e0..7 = kB..kB+7, e8..15 = kB+16..kB+23, kB = k0 + 8*(lane/16).
static __device__ __forceinline__ v16h b_frag_f16(const _Float16* w, int ld,
                                                  int n0, int k0, int lane) {
  int n  = n0 + (lane & 15);
  int kB = k0 + 8 * (lane >> 4);
  const _Float16* p = w + (size_t)n * ld + kB;
  v8h lo = *(const v8h*)(p);
  v8h hi = *(const v8h*)(p + 16);
  v16h r;
#pragma unroll
  for (int i = 0; i < 8; ++i) {
    r[i]     = lo[i];
    r[8 + i] = hi[i];
  }
  return r;
}

// Async global->LDS 16-byte copy (CDNA5; tracked by ASYNCcnt).
static __device__ __forceinline__ void async_g2l_b128(unsigned lds_off,
                                                      const void* gptr) {
  asm volatile("global_load_async_to_lds_b128 %0, %1, off"
               :: "v"(lds_off), "v"((unsigned long long)(uintptr_t)gptr)
               : "memory");
}
static __device__ __forceinline__ void wait_async0() {
  asm volatile("s_wait_asynccnt 0" ::: "memory");
}

// ============ Kernel 0: fp32 -> f16 conversion (one-time) ============
__global__ void cvt_f32_f16_kernel(const float* __restrict__ src,
                                   _Float16* __restrict__ dst, int n4) {
  int i = blockIdx.x * blockDim.x + threadIdx.x;
  if (i < n4) {
    v4f v = *(const v4f*)(src + (size_t)i * 4);
    v4h h;
#pragma unroll
    for (int j = 0; j < 4; ++j) h[j] = (_Float16)v[j];
    *(v4h*)(dst + (size_t)i * 4) = h;
  }
}

// ============ Kernel 1: QKV projection ============
// qkv = xh @ wh^T + b; one wave per 16(M) x 64(N) tile. K-loop is a 2x
// unrolled software pipeline with two alternating fragment sets (no
// register-rotation copies). 64-wide feature tile = one section, one head.
__global__ void __launch_bounds__(256)
mha_qkv_kernel(const _Float16* __restrict__ xh, const _Float16* __restrict__ wh,
               const float* __restrict__ bqkv, _Float16* __restrict__ q,
               _Float16* __restrict__ k, _Float16* __restrict__ vt) {
  int lane = threadIdx.x & 31;
  int wave = blockIdx.x * (blockDim.x >> 5) + (threadIdx.x >> 5);
  const int ET = (3 * D_EMBED) / 64;  // 36
  int m0 = (wave / ET) * 16;
  int et = wave % ET;
  int e0 = et * 64;

  v8f acc[4] = {vzero8(), vzero8(), vzero8(), vzero8()};
  v16h a0, a1, b0[4], b1[4];
  a0 = a_frag_f16(xh, D_EMBED, m0, 0, lane);
#pragma unroll
  for (int c = 0; c < 4; ++c)
    b0[c] = b_frag_f16(wh, D_EMBED, e0 + c * 16, 0, lane);

  for (int k0 = 0; k0 < D_EMBED - 64; k0 += 64) {
    a1 = a_frag_f16(xh, D_EMBED, m0, k0 + 32, lane);
#pragma unroll
    for (int c = 0; c < 4; ++c)
      b1[c] = b_frag_f16(wh, D_EMBED, e0 + c * 16, k0 + 32, lane);
    __builtin_prefetch(xh + (size_t)(m0 + (lane & 15)) * D_EMBED + k0 + 64, 0, 0);
#pragma unroll
    for (int c = 0; c < 4; ++c) acc[c] = wmma_f16(a0, b0[c], acc[c]);

    a0 = a_frag_f16(xh, D_EMBED, m0, k0 + 64, lane);
#pragma unroll
    for (int c = 0; c < 4; ++c)
      b0[c] = b_frag_f16(wh, D_EMBED, e0 + c * 16, k0 + 64, lane);
#pragma unroll
    for (int c = 0; c < 4; ++c) acc[c] = wmma_f16(a1, b1[c], acc[c]);
  }
  // tail: a0/b0 hold k = D-64; last chunk k = D-32
  a1 = a_frag_f16(xh, D_EMBED, m0, D_EMBED - 32, lane);
#pragma unroll
  for (int c = 0; c < 4; ++c)
    b1[c] = b_frag_f16(wh, D_EMBED, e0 + c * 16, D_EMBED - 32, lane);
#pragma unroll
  for (int c = 0; c < 4; ++c) acc[c] = wmma_f16(a0, b0[c], acc[c]);
#pragma unroll
  for (int c = 0; c < 4; ++c) acc[c] = wmma_f16(a1, b1[c], acc[c]);

  int n    = lane & 15;
  int hi   = lane >> 4;
  int sect = et / 12;            // uniform: q/k/v
  int h    = et % 12;            // uniform: head

#pragma unroll
  for (int c = 0; c < 4; ++c) {
    int dd = c * 16 + n;
    float bias = bqkv[e0 + dd];
#pragma unroll
    for (int r = 0; r < 8; ++r) {
      int srow = m0 + r + 8 * hi;      // [0, B*S)
      int b    = srow >> 11;           // SQ = 2048
      int s    = srow & (SQ - 1);
      size_t bh = (size_t)b * N_HEADS + h;
      float val = acc[c][r] + bias;
      if (sect == 0) {
        q[(bh * SQ + s) * HD + dd] = (_Float16)(val * 0.125f);
      } else if (sect == 1) {
        k[(bh * SQ + s) * HD + dd] = (_Float16)val;
      } else {
        vt[(bh * HD + dd) * SQ + s] = (_Float16)val;
      }
    }
  }
}

// ============ Kernel 2: flash attention ============
// Block = 8 waves, same (b,h), consecutive 16-row q tiles. K/V 64x64 tiles
// double-buffered: async-staged into LDS for block kb+64 while computing kb.
__global__ void __launch_bounds__(256)
mha_attn_kernel(const _Float16* __restrict__ q, const _Float16* __restrict__ k,
                const _Float16* __restrict__ vt, _Float16* __restrict__ o) {
  extern __shared__ _Float16 smem[];  // dyn LDS only => offsets start at 0
  // byte layout: K0 @0, K1 @8192, V0 @16384, V1 @24576, P @32768+wwid*2048
  int lane = threadIdx.x & 31;
  int wwid = threadIdx.x >> 5;
  _Float16* pbuf = smem + 16384 + wwid * 1024;  // [16][64] per wave

  int bh = blockIdx.x >> 4;  // 0..23
  int q0 = (((blockIdx.x & 15) << 3) + wwid) << 4;
  int b  = bh / N_HEADS;
  int h  = bh % N_HEADS;

  const _Float16* qh = q  + (size_t)bh * SQ * HD;
  const _Float16* kh = k  + (size_t)bh * SQ * HD;
  const _Float16* vh = vt + (size_t)bh * HD * SQ;
  int n  = lane & 15;
  int hi = lane >> 4;

  v16h qa0 = a_frag_f16(qh, HD, q0, 0,  lane);
  v16h qa1 = a_frag_f16(qh, HD, q0, 32, lane);

  v8f oacc[4] = {vzero8(), vzero8(), vzero8(), vzero8()};
  float run_max[8], run_sum[8];
#pragma unroll
  for (int r = 0; r < 8; ++r) {
    run_max[r] = -__builtin_inff();
    run_sum[r] = 0.f;
  }

  // cooperative async stage of one 64-key block into buffer `bufsel`
  auto stage = [&](int kb, unsigned bufsel) {
    unsigned offK = bufsel * 8192u;
    unsigned offV = 16384u + bufsel * 8192u;
    int c0 = threadIdx.x * 2;
#pragma unroll
    for (int i = 0; i < 2; ++i) {
      int c    = c0 + i;        // 0..511
      int row  = c >> 3;        // 0..63
      int col8 = (c & 7) * 8;   // f16 offset within row
      async_g2l_b128(offK + (unsigned)(c * 16),
                     kh + (size_t)(kb + row) * HD + col8);
      async_g2l_b128(offV + (unsigned)(c * 16),
                     vh + (size_t)row * SQ + kb + col8);
    }
  };

  stage(0, 0);
  const int NBLK = SQ / 64;  // 32
  for (int i = 0; i < NBLK; ++i) {
    int cur = i & 1;
    wait_async0();    // my async writes into buf[cur] done
    __syncthreads();  // everyone's staging done; prev block's reads done
    if (i + 1 < NBLK) stage((i + 1) * 64, 1 - cur);

    const _Float16* smK = smem + (unsigned)cur * 4096;          // [64k][64hd]
    const _Float16* smV = smem + 8192 + (unsigned)cur * 4096;   // [64hd][64k]

    // ---- scores: 4 key tiles of 16, K-dim = hd = 64 (2 chained WMMAs)
    v8f s[4];
#pragma unroll
    for (int t = 0; t < 4; ++t) {
      v8f z = vzero8();
      z = wmma_f16(qa0, b_frag_f16(smK, HD, t * 16, 0,  lane), z);
      s[t] = wmma_f16(qa1, b_frag_f16(smK, HD, t * 16, 32, lane), z);
    }

    // ---- online softmax over the 64-key block
#pragma unroll
    for (int r = 0; r < 8; ++r) {
      float m = fmaxf(fmaxf(s[0][r], s[1][r]), fmaxf(s[2][r], s[3][r]));
      m = fmaxf(m, __shfl_xor(m, 1, 16));
      m = fmaxf(m, __shfl_xor(m, 2, 16));
      m = fmaxf(m, __shfl_xor(m, 4, 16));
      m = fmaxf(m, __shfl_xor(m, 8, 16));
      float nm   = fmaxf(run_max[r], m);
      float corr = __expf(run_max[r] - nm);
      float e0 = __expf(s[0][r] - nm);
      float e1 = __expf(s[1][r] - nm);
      float e2 = __expf(s[2][r] - nm);
      float e3 = __expf(s[3][r] - nm);
      float ps = (e0 + e1) + (e2 + e3);
      ps += __shfl_xor(ps, 1, 16);
      ps += __shfl_xor(ps, 2, 16);
      ps += __shfl_xor(ps, 4, 16);
      ps += __shfl_xor(ps, 8, 16);
      run_sum[r] = run_sum[r] * corr + ps;
      run_max[r] = nm;
      oacc[0][r] *= corr;
      oacc[1][r] *= corr;
      oacc[2][r] *= corr;
      oacc[3][r] *= corr;
      int row = r + 8 * hi;
      pbuf[row * 64 + n]      = (_Float16)e0;
      pbuf[row * 64 + 16 + n] = (_Float16)e1;
      pbuf[row * 64 + 32 + n] = (_Float16)e2;
      pbuf[row * 64 + 48 + n] = (_Float16)e3;
    }

    // ---- P as A fragments (same-wave DS ordering is guaranteed)
    v16h pa0 = a_frag_f16(pbuf, 64, 0, 0,  lane);
    v16h pa1 = a_frag_f16(pbuf, 64, 0, 32, lane);

    // ---- O += P(16x64) * V(64xhd); V tile rows are hd dims
#pragma unroll
    for (int c = 0; c < 4; ++c) {
      oacc[c] = wmma_f16(pa0, b_frag_f16(smV, 64, c * 16, 0,  lane), oacc[c]);
      oacc[c] = wmma_f16(pa1, b_frag_f16(smV, 64, c * 16, 32, lane), oacc[c]);
    }
  }

  // ---- normalize, store [B,S,D] f16 (D index = h*HD + c*16 + n)
#pragma unroll
  for (int c = 0; c < 4; ++c) {
#pragma unroll
    for (int r = 0; r < 8; ++r) {
      int s = q0 + r + 8 * hi;
      float val = oacc[c][r] / run_sum[r];
      o[((size_t)b * SQ + s) * D_EMBED + h * HD + c * 16 + n] = (_Float16)val;
    }
  }
}

// ============ Kernel 3: output projection ============
// out = ao @ w_out^T + b_out (fp32). One wave per 16x64 tile; 2x unrolled
// software pipeline, no rotation copies.
__global__ void __launch_bounds__(256)
mha_proj_kernel(const _Float16* __restrict__ a, const _Float16* __restrict__ wh,
                const float* __restrict__ bias, float* __restrict__ out) {
  int lane = threadIdx.x & 31;
  int wave = blockIdx.x * (blockDim.x >> 5) + (threadIdx.x >> 5);
  const int OT = D_EMBED / 64;  // 12
  int m0 = (wave / OT) * 16;
  int o0 = (wave % OT) * 64;

  v8f acc[4] = {vzero8(), vzero8(), vzero8(), vzero8()};
  v16h a0, a1, b0[4], b1[4];
  a0 = a_frag_f16(a, D_EMBED, m0, 0, lane);
#pragma unroll
  for (int c = 0; c < 4; ++c)
    b0[c] = b_frag_f16(wh, D_EMBED, o0 + c * 16, 0, lane);

  for (int k0 = 0; k0 < D_EMBED - 64; k0 += 64) {
    a1 = a_frag_f16(a, D_EMBED, m0, k0 + 32, lane);
#pragma unroll
    for (int c = 0; c < 4; ++c)
      b1[c] = b_frag_f16(wh, D_EMBED, o0 + c * 16, k0 + 32, lane);
#pragma unroll
    for (int c = 0; c < 4; ++c) acc[c] = wmma_f16(a0, b0[c], acc[c]);

    a0 = a_frag_f16(a, D_EMBED, m0, k0 + 64, lane);
#pragma unroll
    for (int c = 0; c < 4; ++c)
      b0[c] = b_frag_f16(wh, D_EMBED, o0 + c * 16, k0 + 64, lane);
#pragma unroll
    for (int c = 0; c < 4; ++c) acc[c] = wmma_f16(a1, b1[c], acc[c]);
  }
  a1 = a_frag_f16(a, D_EMBED, m0, D_EMBED - 32, lane);
#pragma unroll
  for (int c = 0; c < 4; ++c)
    b1[c] = b_frag_f16(wh, D_EMBED, o0 + c * 16, D_EMBED - 32, lane);
#pragma unroll
  for (int c = 0; c < 4; ++c) acc[c] = wmma_f16(a0, b0[c], acc[c]);
#pragma unroll
  for (int c = 0; c < 4; ++c) acc[c] = wmma_f16(a1, b1[c], acc[c]);

  int n  = lane & 15;
  int hi = lane >> 4;
#pragma unroll
  for (int c = 0; c < 4; ++c) {
    float bv = bias[o0 + c * 16 + n];
#pragma unroll
    for (int r = 0; r < 8; ++r) {
      int row = m0 + r + 8 * hi;
      out[(size_t)row * D_EMBED + o0 + c * 16 + n] = acc[c][r] + bv;
    }
  }
}

extern "C" void kernel_launch(void* const* d_in, const int* in_sizes, int n_in,
                              void* d_out, int out_size, void* d_ws,
                              size_t ws_size, hipStream_t stream) {
  (void)in_sizes; (void)n_in; (void)out_size; (void)ws_size;
  const float* x     = (const float*)d_in[0];
  const float* w_qkv = (const float*)d_in[1];
  const float* b_qkv = (const float*)d_in[2];
  const float* w_out = (const float*)d_in[3];
  const float* b_out = (const float*)d_in[4];
  float* out = (float*)d_out;

  const size_t NX  = (size_t)NB * SQ * D_EMBED;        // 3,145,728
  const size_t NW1 = (size_t)3 * D_EMBED * D_EMBED;    // 1,769,472
  const size_t NW2 = (size_t)D_EMBED * D_EMBED;        //   589,824
  const size_t NQ  = (size_t)NB * N_HEADS * SQ * HD;   // 3,145,728

  _Float16* xh  = (_Float16*)d_ws;
  _Float16* wh1 = xh + NX;
  _Float16* wh2 = wh1 + NW1;
  _Float16* qb  = wh2 + NW2;
  _Float16* kb  = qb + NQ;
  _Float16* vtb = kb + NQ;
  _Float16* ao  = vtb + NQ;   // attention output [B,S,D] f16

  // one-time f16 conversions
  cvt_f32_f16_kernel<<<(int)((NX / 4 + 255) / 256), 256, 0, stream>>>(
      x, xh, (int)(NX / 4));
  cvt_f32_f16_kernel<<<(int)((NW1 / 4 + 255) / 256), 256, 0, stream>>>(
      w_qkv, wh1, (int)(NW1 / 4));
  cvt_f32_f16_kernel<<<(int)((NW2 / 4 + 255) / 256), 256, 0, stream>>>(
      w_out, wh2, (int)(NW2 / 4));

  // K1: 256 * 36 = 9216 waves -> 1152 blocks of 8 waves
  mha_qkv_kernel<<<1152, 256, 0, stream>>>(xh, wh1, b_qkv, qb, kb, vtb);
  // K2: 24 bh * 16 q-chunks = 384 blocks; 48 KB dynamic LDS (double-buffered)
  mha_attn_kernel<<<384, 256, 49152, stream>>>(qb, kb, vtb, ao);
  // K3: 256 * 12 = 3072 waves -> 384 blocks
  mha_proj_kernel<<<384, 256, 0, stream>>>(ao, wh2, b_out, out);
}
